// Quanv_79216376807947
// MI455X (gfx1250) — matrix operand order repaired and checked by reading, not compiled, not run on hardware
//
#include <hip/hip_runtime.h>

typedef __attribute__((ext_vector_type(16))) _Float16 v16h;
typedef __attribute__((ext_vector_type(8)))  float    v8f;

#define HO 112
#define WO 112
#define HW (HO*WO)         // 12544
#define NBATCH 64
#define NPATCH (NBATCH*HW) // 802816 (divisible by 16)
#define WAVES 8
#define PI4 0.78539816339744830962f  // pi/4 : half-angle scale after tanh*pi/2

// ---------------------------------------------------------------------------
// Kernel 1: build the 4 real-symmetric observables M_i = Re(U^dag D_i U)
// from the 24 circuit weights. One wave; lane j (j<16) owns column j of U.
// Output: Mh[0..1023]  = M_i[j][k] as f16  (i*256 + j*16 + k)
//         Mh[1024..2047] = zeros (not read by quanv_kernel)
// ---------------------------------------------------------------------------
__global__ void build_M_kernel(const float* __restrict__ w,
                               _Float16* __restrict__ Mh) {
  __shared__ float Ur[16][16];
  __shared__ float Ui[16][16];
  const int lane = threadIdx.x & 31;
  const int j = lane & 15;

  float ur[16], ui[16];
#pragma unroll
  for (int k = 0; k < 16; ++k) { ur[k] = (k == j) ? 1.f : 0.f; ui[k] = 0.f; }

#pragma unroll
  for (int l = 0; l < 2; ++l) {
    // Rot(phi,theta,omega) on each wire (wire 0 = MSB = bit mask 8)
#pragma unroll
    for (int q = 0; q < 4; ++q) {
      float phi = w[(l*4 + q)*3 + 0];
      float th  = w[(l*4 + q)*3 + 1];
      float om  = w[(l*4 + q)*3 + 2];
      float a  = 0.5f*(phi + om);
      float bb = 0.5f*(phi - om);
      float st, ct, sa, ca, sb, cb;
      __sincosf(0.5f*th, &st, &ct);
      __sincosf(a,  &sa, &ca);
      __sincosf(bb, &sb, &cb);
      // G00=e^{-ia}ct  G01=-e^{ib}st  G10=e^{-ib}st  G11=e^{ia}ct
      float g00r =  ca*ct, g00i = -sa*ct;
      float g01r = -cb*st, g01i = -sb*st;
      float g10r =  cb*st, g10i = -sb*st;
      float g11r =  ca*ct, g11i =  sa*ct;
      const int m = 8 >> q;
#pragma unroll
      for (int k = 0; k < 16; ++k) {
        if (k & m) continue;
        const int k1 = k | m;
        float xr = ur[k],  xi = ui[k];
        float yr = ur[k1], yi = ui[k1];
        ur[k]  = g00r*xr - g00i*xi + g01r*yr - g01i*yi;
        ui[k]  = g00r*xi + g00i*xr + g01r*yi + g01i*yr;
        ur[k1] = g10r*xr - g10i*xi + g11r*yr - g11i*yi;
        ui[k1] = g10r*xi + g10i*xr + g11r*yi + g11i*yr;
      }
    }
    // ring of CNOTs, range r = (l % 3) + 1 ; involutive row permutation
    const int r = (l % 3) + 1;
#pragma unroll
    for (int q = 0; q < 4; ++q) {
      const int mc = 8 >> q;
      const int mt = 8 >> ((q + r) & 3);
      float nr[16], ni[16];
#pragma unroll
      for (int k = 0; k < 16; ++k) {
        const int src = (k & mc) ? (k ^ mt) : k;
        nr[k] = ur[src]; ni[k] = ui[src];
      }
#pragma unroll
      for (int k = 0; k < 16; ++k) { ur[k] = nr[k]; ui[k] = ni[k]; }
    }
  }

  if (lane < 16) {
#pragma unroll
    for (int s = 0; s < 16; ++s) { Ur[s][j] = ur[s]; Ui[s][j] = ui[s]; }
  }
  __syncthreads();

  // M_i[j][k] = sum_s sign_i(s) * Re(conj(U[s][j]) U[s][k]); wire i bit = 3-i
  for (int idx = threadIdx.x; idx < 1024; idx += blockDim.x) {
    const int i  = idx >> 8;
    const int jj = (idx >> 4) & 15;
    const int kk = idx & 15;
    const int bitpos = 3 - i;
    float acc = 0.f;
#pragma unroll
    for (int s = 0; s < 16; ++s) {
      float sign = ((s >> bitpos) & 1) ? -1.f : 1.f;
      acc += sign * (Ur[s][jj]*Ur[s][kk] + Ui[s][jj]*Ui[s][kk]);
    }
    Mh[idx]        = (_Float16)acc;
    Mh[1024 + idx] = (_Float16)0.0f;   // zero region, not read by quanv_kernel
  }
}

// ---------------------------------------------------------------------------
// Kernel 2: per wave, 16 patches -> A tile (16x32 f16, K zero-padded),
// 4 WMMAs against loop-invariant M_i B tiles, LDS transpose, quadratic-form
// epilogue, coalesced stores. 112 % 16 == 0 -> tiles never straddle rows.
// ---------------------------------------------------------------------------
__global__ __launch_bounds__(256) void quanv_kernel(
    const float* __restrict__ x, const _Float16* __restrict__ Mh,
    float* __restrict__ out) {
  __shared__ float Cs[WAVES][4][16][16];

  const int wave = threadIdx.x >> 5;
  const int lane = threadIdx.x & 31;
  const int hi   = lane >> 4;       // 0: A rows K=0..7, 1: K=8..15
  const int pl   = lane & 15;       // patch-in-tile / B column

  const int tile = blockIdx.x * WAVES + wave;
  const int p    = tile * 16 + pl;
  const int wo   = p % WO;
  const int t1   = p / WO;
  const int ho   = t1 % HO;
  const int b    = t1 / HO;

  // ---- loop-invariant B tiles: lanes 0-15 read M row K=0..7, lanes 16-31
  //      K=8..15 (symmetric = cols); elements 8..15 are K=16..31 zero pad ----
  const _Float16* Mbase = Mh + (hi ? 8 : 0) + pl*16;
  v16h Bv[4];
#pragma unroll
  for (int i = 0; i < 4; ++i) {
#pragma unroll
    for (int e = 0; e < 16; ++e)
      Bv[i][e] = (e < 8) ? Mbase[i*256 + e] : (_Float16)0.0f;   // contiguous 16B -> b128 load
  }

  // ---- patch pixels: x[b,0,2ho..2ho+1, 2wo..2wo+1] (two float2 loads) ----
  const float2* x2 = (const float2*)x;
  float2 r0 = x2[(b*224 + 2*ho    )*112 + wo];
  float2 r1 = x2[(b*224 + 2*ho + 1)*112 + wo];

  float c0,s0,c1,s1,c2,s2,c3,s3;
  __sincosf(tanhf(r0.x)*PI4, &s0, &c0);
  __sincosf(tanhf(r0.y)*PI4, &s1, &c1);
  __sincosf(tanhf(r1.x)*PI4, &s2, &c2);
  __sincosf(tanhf(r1.y)*PI4, &s3, &c3);

  // t[e] = prod over wires 1..3 ; psi0[k] = (k&8 ? s0 : c0) * t[k&7]
  float t23[4];
  t23[0] = c2*c3; t23[1] = c2*s3; t23[2] = s2*c3; t23[3] = s2*s3;
  float t[8];
#pragma unroll
  for (int e = 0; e < 8; ++e)
    t[e] = ((e & 4) ? s1 : c1) * t23[e & 3];

  float psi[16];
#pragma unroll
  for (int k = 0; k < 16; ++k)
    psi[k] = ((k & 8) ? s0 : c0) * t[k & 7];   // constant indices only

  // ---- A tile: single select f0, then constant-index products ----
  const float f0 = hi ? s0 : c0;               // lane half picks K block
  v16h Av;
#pragma unroll
  for (int e = 0; e < 8; ++e)
    Av[e] = (_Float16)(f0 * t[e]);
#pragma unroll
  for (int e = 8; e < 16; ++e)
    Av[e] = (_Float16)0.0f;                    // K=16..31 pad

  // ---- 4 WMMAs: C_i[p][n] = sum_k psi_p[k] * M_i[k][n] ----
  v8f zero = {};
#pragma unroll
  for (int i = 0; i < 4; ++i) {
    v8f acc = __builtin_amdgcn_wmma_f32_16x16x32_f16(
        false, Av, false, Bv[i], (short)0, zero, false, false);
#pragma unroll
    for (int r = 0; r < 8; ++r)
      Cs[wave][i][hi*8 + r][pl] = acc[r];      // C layout: row hi*8+r, col pl
  }
  __syncthreads();

  // ---- epilogue: z_i[p] = sum_k psi_p[k] * C_i[p][k] ----
  const int ia = hi ? 2 : 0;  // lanes 0-15 -> i=0,1 ; lanes 16-31 -> i=2,3
  float z0 = 0.f, z1 = 0.f;
#pragma unroll
  for (int k = 0; k < 16; ++k) {
    z0 += psi[k] * Cs[wave][ia    ][pl][k];    // contiguous 64B -> b128 loads
    z1 += psi[k] * Cs[wave][ia + 1][pl][k];
  }
  // out shape (b, i, ho, wo)
  const long base = (((long)b*4 + ia)*HO + ho)*WO + wo;
  out[base]      = z0;
  out[base + HW] = z1;
}

// ---------------------------------------------------------------------------
extern "C" void kernel_launch(void* const* d_in, const int* in_sizes, int n_in,
                              void* d_out, int out_size, void* d_ws, size_t ws_size,
                              hipStream_t stream) {
  const float* x = (const float*)d_in[0];        // (64,1,224,224) f32
  const float* w = (const float*)d_in[1];        // (2,4,3) f32
  float* out = (float*)d_out;                    // (64,4,112,112) f32
  _Float16* Mh = (_Float16*)d_ws;                // 2048 halves (4KB scratch)

  build_M_kernel<<<1, 32, 0, stream>>>(w, Mh);

  const int tiles  = NPATCH / 16;                // 50176
  const int blocks = tiles / WAVES;              // 6272 (exact)
  quanv_kernel<<<blocks, 256, 0, stream>>>(x, Mh, out);
}
